// SimpleMoeBlock_27367531610987
// MI455X (gfx1250) — matrix operand order repaired and checked by reading, not compiled, not run on hardware
//
#include <hip/hip_runtime.h>
#include <hip/hip_bf16.h>

// Problem sizes (fixed by the reference)
#define HD     512      // hidden size
#define ID     1024     // intermediate size
#define NE     32       // experts
#define NT     8192     // tokens = 2*4096
#define MT     64       // tokens per tile (4 WMMA row-blocks)
#define CAP    8192     // per-expert list capacity (== NT, cannot overflow)

typedef __attribute__((ext_vector_type(16))) __bf16 v16bf;
typedef __attribute__((ext_vector_type(8)))  float  v8f;

// round-to-nearest-even f32 -> bf16 (bit pattern)
__device__ __forceinline__ unsigned short f2bf(float f) {
  unsigned u = __float_as_uint(f);
  u += 0x7FFFu + ((u >> 16) & 1u);
  return (unsigned short)(u >> 16);
}

__device__ __forceinline__ v8f wmma_bf16(v16bf a, v16bf b, v8f c) {
  // (neg_a, A, neg_b, B, c_mod, C, reuse_a, reuse_b)
  return __builtin_amdgcn_wmma_f32_16x16x32_bf16(false, a, false, b, (short)0, c,
                                                 false, false);
}

// A fragment (16x32 bf16): p = row + (k0 + half*8); chunks at +0 and +16 elems.
// Per ISA: lanes 0-15 hold K 0-7 & 16-23; lanes 16-31 hold K 8-15 & 24-31.
__device__ __forceinline__ v16bf load_frag_a(const unsigned short* p) {
  union { uint4 q[2]; v16bf v; } f;
  f.q[0] = *(const uint4*)(p);
  f.q[1] = *(const uint4*)(p + 16);
  return f.v;
}

// B fragment (32x16 bf16) from column-major (transposed) weights:
// p = colPtr + k0 + half*8; chunks at +0 and +16 elems (same K order as A).
__device__ __forceinline__ v16bf load_frag_b(const unsigned short* p) {
  union { uint4 q[2]; v16bf v; } f;
  f.q[0] = *(const uint4*)(p);
  f.q[1] = *(const uint4*)(p + 16);
  return f.v;
}

// fast silu: g * rcp(1+exp(-g)); result is truncated to bf16 anyway
__device__ __forceinline__ float fast_silu(float g) {
  return g * __builtin_amdgcn_rcpf(1.0f + __expf(-g));
}

// ---------------------------------------------------------------- zero kernels
__global__ void zero_counts_kernel(int* counts) {
  if (threadIdx.x < NE) counts[threadIdx.x] = 0;
}

__global__ void zero_out_kernel(float4* out) {
  out[(size_t)blockIdx.x * blockDim.x + threadIdx.x] = float4{0.f, 0.f, 0.f, 0.f};
}

// ---------------------------------------------------------------- pre-pass
// x (f32 row-major) -> bf16 row-major, packed pair stores
__global__ __launch_bounds__(256) void convert_x_kernel(
    const float2* __restrict__ in, unsigned* __restrict__ out) {
  size_t i = (size_t)blockIdx.x * 256 + threadIdx.x;
  float2 v = in[i];
  out[i] = (unsigned)f2bf(v.x) | ((unsigned)f2bf(v.y) << 16);
}

// per-slice transpose + f32->bf16: in[z][r][c] -> out[z][c][r]
__global__ __launch_bounds__(256) void transpose_bf16_kernel(
    const float* __restrict__ in, unsigned short* __restrict__ out,
    int R, int C) {
  __shared__ unsigned short tile[32][33];
  const int e  = blockIdx.z;
  const float* ip = in + (size_t)e * R * C;
  unsigned short* op = out + (size_t)e * R * C;
  const int r0 = blockIdx.y * 32, c0 = blockIdx.x * 32;
  const int tid = threadIdx.x;
  for (int i = tid; i < 1024; i += 256) {
    int r = i >> 5, c = i & 31;
    tile[r][c] = f2bf(ip[(size_t)(r0 + r) * C + c0 + c]);
  }
  __syncthreads();
  for (int i = tid; i < 512; i += 256) {          // packed uint stores, coalesced
    int c = i >> 4, rp = (i & 15) * 2;
    unsigned lo = tile[rp][c], hi = tile[rp + 1][c];
    *(unsigned*)(op + (size_t)(c0 + c) * R + r0 + rp) = lo | (hi << 16);
  }
}

// ---------------------------------------------------------------- router (WMMA)
// Block = 256 threads (8 waves x 16 tokens = 128 tokens). Gate GEMM on the
// matrix pipe: 2 column tiles (experts 0-15 / 16-31), f32 logits -> padded LDS,
// then 128 threads do softmax + top-2 + atomic append.
__global__ __launch_bounds__(256) void router_wmma_kernel(
    const unsigned short* __restrict__ xbf,
    const unsigned short* __restrict__ WgateT,   // [32][512] bf16
    const float* __restrict__ bgate, int* __restrict__ counts,
    int* __restrict__ alist, float* __restrict__ wl) {
  __shared__ float lg[128][33];                  // stride 33: bank-conflict-free
  const int tid  = threadIdx.x;
  const int wv   = tid >> 5;
  const int lane = tid & 31;
  const int lrow = lane & 15;
  const int half = lane >> 4;
  const int t0   = blockIdx.x * 128;

  const unsigned short* arow = xbf + (size_t)(t0 + wv * 16 + lrow) * HD;
  v8f a0, a1;
  {
    const float b0 = bgate[lrow], b1 = bgate[lrow + 16];
#pragma unroll
    for (int j = 0; j < 8; ++j) { a0[j] = b0; a1[j] = b1; }
  }
  const unsigned short* bp0 = WgateT + (size_t)lrow * HD;
  const unsigned short* bp1 = WgateT + (size_t)(lrow + 16) * HD;
#pragma unroll 4
  for (int k0 = 0; k0 < HD; k0 += 32) {
    v16bf a  = load_frag_a(arow + k0 + half * 8);
    v16bf b0 = load_frag_b(bp0 + k0 + half * 8);
    v16bf b1 = load_frag_b(bp1 + k0 + half * 8);
    a0 = wmma_bf16(a, b0, a0);
    a1 = wmma_bf16(a, b1, a1);
  }
#pragma unroll
  for (int j = 0; j < 8; ++j) {
    int row = wv * 16 + j + 8 * half;            // C layout: M = j + 8*half
    lg[row][lrow]      = a0[j];                  // experts 0-15  (N = lane)
    lg[row][lrow + 16] = a1[j];                  // experts 16-31
  }
  __syncthreads();

  if (tid < 128) {
    const int t = t0 + tid;
    float logit[NE];
#pragma unroll
    for (int e = 0; e < NE; ++e) logit[e] = lg[tid][e];
    float m = logit[0];
#pragma unroll
    for (int e = 1; e < NE; ++e) m = fmaxf(m, logit[e]);
    float s = 0.f;
#pragma unroll
    for (int e = 0; e < NE; ++e) { logit[e] = __expf(logit[e] - m); s += logit[e]; }
    float inv = __builtin_amdgcn_rcpf(s);
    int i1 = 0; float p1 = logit[0];
#pragma unroll
    for (int e = 1; e < NE; ++e) if (logit[e] > p1) { p1 = logit[e]; i1 = e; }
    int i2 = (i1 == 0) ? 1 : 0; float p2 = logit[i2];
#pragma unroll
    for (int e = 0; e < NE; ++e)
      if (e != i1 && logit[e] > p2) { p2 = logit[e]; i2 = e; }
    int s1 = atomicAdd(&counts[i1], 1);
    alist[i1 * CAP + s1] = t * 2;     wl[i1 * CAP + s1] = p1 * inv;
    int s2 = atomicAdd(&counts[i2], 1);
    alist[i2 * CAP + s2] = t * 2 + 1; wl[i2 * CAP + s2] = p2 * inv;
  }
}

// ---------------------------------------------------------------- fused expert
// Block: 256 threads (8 wave32 = 4 row-blocks x 2 col-pairs), 64 tokens of one expert.
// Each wave owns TWO 16-col tiles -> one A fragment feeds 4 (phase1) / 2 (phase2)
// WMMAs, halving A traffic and giving independent accumulation chains.
// Phase 1: h = silu(x*Wg+bg)*(x*Wu+bu) -> global bf16 hbuf[assignment][I]
// Phase 2: out += w * (h*Wd + bd)      -> global f32 atomics
__global__ __launch_bounds__(256) void moe_expert_kernel(
    const unsigned short* __restrict__ xbf,
    const unsigned short* __restrict__ WgT, const float* __restrict__ bg,
    const unsigned short* __restrict__ WuT, const float* __restrict__ bu,
    const unsigned short* __restrict__ WdT, const float* __restrict__ bd,
    const int* __restrict__ counts, const int* __restrict__ alist,
    const float* __restrict__ wl, unsigned short* __restrict__ hbuf,
    float* __restrict__ out) {
  const int e  = blockIdx.y;
  const int mt = blockIdx.x;
  const int count = counts[e];
  if (mt * MT >= count) return;                 // uniform per block

  __shared__ int   a_s[MT];
  __shared__ float w_s[MT];

  const int tid  = threadIdx.x;
  const int wv   = tid >> 5;
  const int lane = tid & 31;
  const int lrow = lane & 15;
  const int half = lane >> 4;
  const int rb   = wv & 3;          // row-block 0..3
  const int cb   = wv >> 2;         // col-pair  0..1 (owns cols cb*32 .. cb*32+31)

  const int valid = min(MT, count - mt * MT);
  if (tid < MT) {
    int slot = mt * MT + tid;
    if (tid < valid) { a_s[tid] = alist[e * CAP + slot]; w_s[tid] = wl[e * CAP + slot]; }
    else             { a_s[tid] = alist[e * CAP + mt * MT]; w_s[tid] = 0.f; }
  }
  __syncthreads();

  const int row16 = rb * 16 + lrow;                        // this lane's A row
  const unsigned short* arow = xbf + (size_t)(a_s[row16] >> 1) * HD;

  // ---------------- Phase 1: h -> hbuf ----------------
  for (int ns = 0; ns < ID / 64; ++ns) {
    const int c0 = ns * 64 + cb * 32 + lrow;   // first 16-col tile
    const int c1 = c0 + 16;                    // second 16-col tile
    v8f ag0, au0, ag1, au1;
    {
      const float bg0 = bg[e * ID + c0], bu0 = bu[e * ID + c0];
      const float bg1 = bg[e * ID + c1], bu1 = bu[e * ID + c1];
#pragma unroll
      for (int j = 0; j < 8; ++j) { ag0[j] = bg0; au0[j] = bu0; ag1[j] = bg1; au1[j] = bu1; }
    }
    const unsigned short* bgp0 = WgT + ((size_t)e * ID + c0) * HD;
    const unsigned short* bup0 = WuT + ((size_t)e * ID + c0) * HD;
    const unsigned short* bgp1 = WgT + ((size_t)e * ID + c1) * HD;
    const unsigned short* bup1 = WuT + ((size_t)e * ID + c1) * HD;
#pragma unroll 2
    for (int k0 = 0; k0 < HD; k0 += 32) {
      v16bf a   = load_frag_a(arow + k0 + half * 8);
      v16bf bG0 = load_frag_b(bgp0 + k0 + half * 8);
      v16bf bU0 = load_frag_b(bup0 + k0 + half * 8);
      v16bf bG1 = load_frag_b(bgp1 + k0 + half * 8);
      v16bf bU1 = load_frag_b(bup1 + k0 + half * 8);
      ag0 = wmma_bf16(a, bG0, ag0);
      au0 = wmma_bf16(a, bU0, au0);
      ag1 = wmma_bf16(a, bG1, ag1);
      au1 = wmma_bf16(a, bU1, au1);
    }
#pragma unroll
    for (int j = 0; j < 8; ++j) {
      int rowj = rb * 16 + j + 8 * half;                  // C layout: M=j+8*half
      unsigned short* hr = hbuf + (size_t)a_s[rowj] * ID;
      hr[c0] = f2bf(fast_silu(ag0[j]) * au0[j]);          // silu(g)*u
      hr[c1] = f2bf(fast_silu(ag1[j]) * au1[j]);
    }
  }
  __threadfence();        // make hbuf stores visible to other waves in the block
  __syncthreads();

  // ---------------- Phase 2: out += w * (h*Wd + bd) ----------------
  const unsigned short* hrow = hbuf + (size_t)a_s[row16] * ID;
  for (int hs = 0; hs < HD / 64; ++hs) {
    const int c0 = hs * 64 + cb * 32 + lrow;
    const int c1 = c0 + 16;
    v8f ac0, ac1;
    {
      const float b0 = bd[e * HD + c0], b1 = bd[e * HD + c1];
#pragma unroll
      for (int j = 0; j < 8; ++j) { ac0[j] = b0; ac1[j] = b1; }
    }
    const unsigned short* bdp0 = WdT + ((size_t)e * HD + c0) * ID;
    const unsigned short* bdp1 = WdT + ((size_t)e * HD + c1) * ID;
#pragma unroll 4
    for (int k0 = 0; k0 < ID; k0 += 32) {
      v16bf a  = load_frag_a(hrow + k0 + half * 8);
      v16bf b0 = load_frag_b(bdp0 + k0 + half * 8);
      v16bf b1 = load_frag_b(bdp1 + k0 + half * 8);
      ac0 = wmma_bf16(a, b0, ac0);
      ac1 = wmma_bf16(a, b1, ac1);
    }
#pragma unroll
    for (int j = 0; j < 8; ++j) {
      int rowj = rb * 16 + j + 8 * half;
      float w  = w_s[rowj];
      if (w != 0.0f) {
        float* orow = out + (size_t)(a_s[rowj] >> 1) * HD;
        atomicAdd(orow + c0, w * ac0[j]);
        atomicAdd(orow + c1, w * ac1[j]);
      }
    }
  }
}

// ---------------------------------------------------------------- launch
extern "C" void kernel_launch(void* const* d_in, const int* in_sizes, int n_in,
                              void* d_out, int out_size, void* d_ws, size_t ws_size,
                              hipStream_t stream) {
  const float* x     = (const float*)d_in[0];   // [2,4096,512]
  const float* Wgate = (const float*)d_in[1];   // [512,32]
  const float* bgate = (const float*)d_in[2];   // [32]
  const float* Wg    = (const float*)d_in[3];   // [32,512,1024]
  const float* bg    = (const float*)d_in[4];   // [32,1024]
  const float* Wu    = (const float*)d_in[5];   // [32,512,1024]
  const float* bu    = (const float*)d_in[6];   // [32,1024]
  const float* Wd    = (const float*)d_in[7];   // [32,1024,512]
  const float* bd    = (const float*)d_in[8];   // [32,512]
  float* out = (float*)d_out;                   // [2,4096,512] f32

  // workspace layout (bytes)
  char* ws = (char*)d_ws;
  size_t o = 0;
  int*            counts = (int*)(ws + o);            o += 256;
  int*            alist  = (int*)(ws + o);            o += (size_t)NE * CAP * 4;
  float*          wl     = (float*)(ws + o);          o += (size_t)NE * CAP * 4;
  unsigned short* xbf    = (unsigned short*)(ws + o); o += (size_t)NT * HD * 2;
  unsigned short* WgateT = (unsigned short*)(ws + o); o += (size_t)NE * HD * 2;
  unsigned short* WgT    = (unsigned short*)(ws + o); o += (size_t)NE * HD * ID * 2;
  unsigned short* WuT    = (unsigned short*)(ws + o); o += (size_t)NE * HD * ID * 2;
  unsigned short* WdT    = (unsigned short*)(ws + o); o += (size_t)NE * HD * ID * 2;
  unsigned short* hbuf   = (unsigned short*)(ws + o); o += (size_t)NT * 2 * ID * 2;

  zero_counts_kernel<<<1, 64, 0, stream>>>(counts);
  zero_out_kernel<<<(NT * HD) / (4 * 256), 256, 0, stream>>>((float4*)out);

  convert_x_kernel<<<(NT * HD / 2) / 256, 256, 0, stream>>>((const float2*)x,
                                                            (unsigned*)xbf);
  {
    dim3 gW(1, HD / 32, 1);           // Wgate: [512][32] -> [32][512]
    transpose_bf16_kernel<<<gW, 256, 0, stream>>>(Wgate, WgateT, HD, NE);
    dim3 gGU(ID / 32, HD / 32, NE);   // Wg/Wu: [512][1024] -> [1024][512]
    transpose_bf16_kernel<<<gGU, 256, 0, stream>>>(Wg, WgT, HD, ID);
    transpose_bf16_kernel<<<gGU, 256, 0, stream>>>(Wu, WuT, HD, ID);
    dim3 gD(HD / 32, ID / 32, NE);    // Wd: [1024][512] -> [512][1024]
    transpose_bf16_kernel<<<gD, 256, 0, stream>>>(Wd, WdT, ID, HD);
  }

  router_wmma_kernel<<<NT / 128, 256, 0, stream>>>(xbf, WgateT, bgate,
                                                   counts, alist, wl);

  dim3 grid(CAP / MT, NE);            // 128 x 32; empty tiles early-exit
  moe_expert_kernel<<<grid, 256, 0, stream>>>(xbf, WgT, bg, WuT, bu, WdT, bd,
                                              counts, alist, wl, hbuf, out);
}
